// ArcFace_49091476193958
// MI455X (gfx1250) — matrix operand order repaired and checked by reading, not compiled, not run on hardware
//
#include <hip/hip_runtime.h>
#include <hip/hip_bf16.h>
#include <math.h>

typedef __bf16 bf16_t;
typedef __attribute__((ext_vector_type(16))) __bf16 v16bf;
typedef __attribute__((ext_vector_type(8)))  float  v8f;

#define S_SCALE 30.0f
#define EPS_LS  0.1f
#define COSM_C  0.8775825618903728f
#define SINM_C  0.479425538604203f
#define TH_C   (-0.8775825618903728f)
#define MM_C    0.2397127693021015f

#define D_DIM 512
#define NBLK  64      // classes per workgroup
#define KT    32      // WMMA K per step

// dynamic LDS layout (bytes)
#define SB_OFF        0                      // [64][40] bf16 (pad 32->40) = 5120
#define WSCALE_OFF    5120                   // 64 f32
#define SRED_OFF      5376                   // 64*4 f32 = 1024
#define LOGITS_OFF    6400                   // 512*68 f32 = 139264
#define LOGITS_STRIDE 68
#define SMEM_BYTES    (6400 + 512 * LOGITS_STRIDE * 4)   // 145664

union FragU { uint4 q[2]; v16bf v; };
union Pack8 { uint4 q; bf16_t h[8]; };

// ---------------- Kernel 1: normalize x rows, emit bf16 ----------------
__global__ void xnorm_kernel(const float* __restrict__ x, bf16_t* __restrict__ xn) {
    __shared__ float red[256];
    const int m = blockIdx.x;
    const int t = threadIdx.x;
    float a = x[(size_t)m * D_DIM + t];
    float b = x[(size_t)m * D_DIM + t + 256];
    red[t] = a * a + b * b;
    __syncthreads();
    #pragma unroll
    for (int s = 128; s > 0; s >>= 1) {
        if (t < s) red[t] += red[t + s];
        __syncthreads();
    }
    const float scale = 1.0f / fmaxf(sqrtf(red[0]), 1e-12f);
    xn[(size_t)m * D_DIM + t]       = (bf16_t)(a * scale);
    xn[(size_t)m * D_DIM + t + 256] = (bf16_t)(b * scale);
}

// ------- Kernel 2: WMMA GEMM + fused weight-norm + margin + online softmax -------
__global__ void __launch_bounds__(256)
arcface_gemm_kernel(const bf16_t* __restrict__ xn, const float* __restrict__ w,
                    const int* __restrict__ label,
                    float* __restrict__ pmax, float* __restrict__ psum,
                    float* __restrict__ psl,  float* __restrict__ tgt,
                    int C, int NB) {
    extern __shared__ char smem[];
    bf16_t* sB     = (bf16_t*)(smem + SB_OFF);
    float*  wscale = (float*)(smem + WSCALE_OFF);
    float*  sred   = (float*)(smem + SRED_OFF);
    float*  slog   = (float*)(smem + LOGITS_OFF);

    const int tid  = threadIdx.x;
    const int blk  = blockIdx.x;
    const int c0   = blk * NBLK;
    const int lane = tid & 31;
    const int wv   = tid >> 5;        // 0..7, each wave owns 64 rows
    const int half = lane >> 4;       // 0/1
    const int l15  = lane & 15;

    // staging map: 4 threads per class row, 8 f32 each
    const int scl = tid >> 2;                 // class_local 0..63
    const int sk  = (tid & 3) * 8;            // k offset within KT
    const int cIdx = min(c0 + scl, C - 1);    // clamp tail
    const float* wrow = w + (size_t)cIdx * D_DIM + sk;

    v8f acc[4][4];
    #pragma unroll
    for (int i = 0; i < 4; ++i)
        #pragma unroll
        for (int j = 0; j < 4; ++j) acc[i][j] = {};

    float ssw = 0.0f;
    const int kbase = half * 8;   // A-fragment K sub-base per lane half

    for (int ks = 0; ks < D_DIM / KT; ++ks) {
        __syncthreads();   // previous tile fully consumed

        // stage 64x32 f32 weight tile -> bf16 LDS, accumulate sum(w^2)
        const float4 f0 = *(const float4*)(wrow + ks * KT);
        const float4 f1 = *(const float4*)(wrow + ks * KT + 4);
        if (ks + 1 < D_DIM / KT) __builtin_prefetch(wrow + (ks + 1) * KT);
        ssw += f0.x * f0.x + f0.y * f0.y + f0.z * f0.z + f0.w * f0.w
             + f1.x * f1.x + f1.y * f1.y + f1.z * f1.z + f1.w * f1.w;
        Pack8 p;
        p.h[0] = (bf16_t)f0.x; p.h[1] = (bf16_t)f0.y;
        p.h[2] = (bf16_t)f0.z; p.h[3] = (bf16_t)f0.w;
        p.h[4] = (bf16_t)f1.x; p.h[5] = (bf16_t)f1.y;
        p.h[6] = (bf16_t)f1.z; p.h[7] = (bf16_t)f1.w;
        *(uint4*)(sB + scl * 40 + sk) = p.q;

        __syncthreads();   // tile visible

        // A fragments: rows wv*64 + rt*16 + l15, K layout per ISA 16-bit A map
        v16bf afrag[4];
        #pragma unroll
        for (int rt = 0; rt < 4; ++rt) {
            const int row = wv * 64 + rt * 16 + l15;
            const bf16_t* pA = xn + (size_t)row * D_DIM + ks * KT + kbase;
            FragU fa;
            fa.q[0] = *(const uint4*)pA;          // K kb..kb+7
            fa.q[1] = *(const uint4*)(pA + 16);   // K kb+16..kb+23
            afrag[rt] = fa.v;
        }
        // B fragments from LDS: col = ct*16 + l15, K kb..kb+7 and kb+16..kb+23 per lane
        v16bf bfrag[4];
        #pragma unroll
        for (int ct = 0; ct < 4; ++ct) {
            const bf16_t* pB = sB + (ct * 16 + l15) * 40 + half * 8;
            FragU fb;
            fb.q[0] = *(const uint4*)pB;
            fb.q[1] = *(const uint4*)(pB + 16);
            bfrag[ct] = fb.v;
        }
        #pragma unroll
        for (int rt = 0; rt < 4; ++rt)
            #pragma unroll
            for (int ct = 0; ct < 4; ++ct)
                acc[rt][ct] = __builtin_amdgcn_wmma_f32_16x16x32_bf16(
                    false, afrag[rt], false, bfrag[ct],
                    (short)0, acc[rt][ct], false, false);
    }

    // per-class sum-of-squares partials (4 threads per class)
    sred[scl * 4 + (tid & 3)] = ssw;

    // dump accumulators to LDS logits[512][68]
    #pragma unroll
    for (int rt = 0; rt < 4; ++rt)
        #pragma unroll
        for (int ct = 0; ct < 4; ++ct)
            #pragma unroll
            for (int r = 0; r < 8; ++r) {
                const int m = wv * 64 + rt * 16 + half * 8 + r;
                slog[m * LOGITS_STRIDE + ct * 16 + l15] = acc[rt][ct][r];
            }
    __syncthreads();

    if (tid < 64) {
        const float s4 = sred[tid * 4] + sred[tid * 4 + 1] +
                         sred[tid * 4 + 2] + sred[tid * 4 + 3];
        wscale[tid] = 1.0f / fmaxf(sqrtf(s4), 1e-12f);
    }
    __syncthreads();

    // online softmax scan: 2 rows per thread over this block's columns
    const int valid = min(NBLK, C - c0);
    #pragma unroll
    for (int rep = 0; rep < 2; ++rep) {
        const int m = tid + rep * 256;
        const int labLoc = label[m] - c0;
        const float* lr = slog + m * LOGITS_STRIDE;
        float mx = -INFINITY, se = 0.0f, sl = 0.0f;
        for (int c = 0; c < valid; ++c) {
            const float cosv = lr[c] * wscale[c];
            float lg = S_SCALE * cosv;
            if (c == labLoc) {
                const float s2  = fminf(fmaxf(1.0f - cosv * cosv, 0.0f), 1.0f);
                const float sin = sqrtf(s2);
                float phi = cosv * COSM_C - sin * SINM_C;
                phi = (cosv > TH_C) ? phi : (cosv - MM_C);
                lg = S_SCALE * phi;
                tgt[m] = lg;          // exactly one block owns the label column
            }
            const float mn = fmaxf(mx, lg);
            se = se * __expf(mx - mn) + __expf(lg - mn);
            mx = mn;
            sl += lg;
        }
        pmax[(size_t)m * NB + blk] = mx;
        psum[(size_t)m * NB + blk] = se;
        psl [(size_t)m * NB + blk] = sl;
    }
}

// ---------------- Kernel 3: combine per-block partials per row ----------------
__global__ void row_reduce_kernel(const float* __restrict__ pmax,
                                  const float* __restrict__ psum,
                                  const float* __restrict__ psl,
                                  const float* __restrict__ tgt,
                                  float* __restrict__ rowloss, int C, int NB) {
    __shared__ float red[256];
    const int m = blockIdx.x, t = threadIdx.x;
    const float* pm = pmax + (size_t)m * NB;
    const float* ps = psum + (size_t)m * NB;
    const float* pl = psl  + (size_t)m * NB;

    float mx = -INFINITY;
    for (int b = t; b < NB; b += 256) mx = fmaxf(mx, pm[b]);
    red[t] = mx; __syncthreads();
    for (int s = 128; s > 0; s >>= 1) { if (t < s) red[t] = fmaxf(red[t], red[t + s]); __syncthreads(); }
    const float M = red[0];
    __syncthreads();

    float se = 0.0f, sl = 0.0f;
    for (int b = t; b < NB; b += 256) { se += ps[b] * __expf(pm[b] - M); sl += pl[b]; }
    red[t] = se; __syncthreads();
    for (int s = 128; s > 0; s >>= 1) { if (t < s) red[t] += red[t + s]; __syncthreads(); }
    se = red[0]; __syncthreads();
    red[t] = sl; __syncthreads();
    for (int s = 128; s > 0; s >>= 1) { if (t < s) red[t] += red[t + s]; __syncthreads(); }
    sl = red[0];

    if (t == 0) {
        const float lse = M + logf(se);
        rowloss[m] = -(1.0f - EPS_LS) * tgt[m] - (EPS_LS / (float)C) * sl + lse;
    }
}

// ---------------- Kernel 4: mean over rows -> scalar loss ----------------
__global__ void final_kernel(const float* __restrict__ rowloss, float* __restrict__ out, int B) {
    __shared__ float red[256];
    const int t = threadIdx.x;
    red[t] = rowloss[t] + rowloss[t + 256];
    __syncthreads();
    for (int s = 128; s > 0; s >>= 1) { if (t < s) red[t] += red[t + s]; __syncthreads(); }
    if (t == 0) out[0] = red[0] / (float)B;
}

extern "C" void kernel_launch(void* const* d_in, const int* in_sizes, int n_in,
                              void* d_out, int out_size, void* d_ws, size_t ws_size,
                              hipStream_t stream) {
    const float* x     = (const float*)d_in[0];
    const float* w     = (const float*)d_in[1];
    const int*   label = (const int*)d_in[2];
    float*       out   = (float*)d_out;

    const int D  = D_DIM;
    const int B  = in_sizes[0] / D;        // 512
    const int C  = in_sizes[1] / D;        // 100000
    const int NB = (C + NBLK - 1) / NBLK;  // 1563

    char* ws = (char*)d_ws;
    bf16_t* xn = (bf16_t*)ws;
    size_t off = (size_t)B * D * sizeof(bf16_t);
    float* pmax    = (float*)(ws + off); off += (size_t)B * NB * sizeof(float);
    float* psum    = (float*)(ws + off); off += (size_t)B * NB * sizeof(float);
    float* psl     = (float*)(ws + off); off += (size_t)B * NB * sizeof(float);
    float* tgt     = (float*)(ws + off); off += (size_t)B * sizeof(float);
    float* rowloss = (float*)(ws + off); off += (size_t)B * sizeof(float);

    xnorm_kernel<<<B, 256, 0, stream>>>(x, xn);
    arcface_gemm_kernel<<<NB, 256, SMEM_BYTES, stream>>>(xn, w, label,
                                                         pmax, psum, psl, tgt, C, NB);
    row_reduce_kernel<<<B, 256, 0, stream>>>(pmax, psum, psl, tgt, rowloss, C, NB);
    final_kernel<<<1, 256, 0, stream>>>(rowloss, out, B);
}
